// ProgramGNN_62732292326129
// MI455X (gfx1250) — matrix operand order repaired and checked by reading, not compiled, not run on hardware
//
#include <hip/hip_runtime.h>
#include <math.h>

#define HEADS 4
#define HD 32
#define HID 128
#define NEG_SLOPE 0.2f
#define LN_EPS 1e-5f

typedef __bf16 bf16_t;
typedef __attribute__((ext_vector_type(16))) __bf16 v16bf;
typedef __attribute__((ext_vector_type(8)))  __bf16 v8bf;
typedef __attribute__((ext_vector_type(8)))  float  v8f;

__device__ __forceinline__ bf16_t f2bf(float f) {
    unsigned u = __float_as_uint(f);
    u += 0x7FFFu + ((u >> 16) & 1u);           // round-to-nearest-even
    unsigned short s = (unsigned short)(u >> 16);
    bf16_t b;
    __builtin_memcpy(&b, &s, 2);
    return b;
}

// float atomic-max via int/uint ordering trick (lowers to native global_atomic_max_i32/min_u32)
__device__ __forceinline__ void atomicMaxF32(float* addr, float v) {
    if (v >= 0.0f) atomicMax((int*)addr, __float_as_int(v));
    else           atomicMin((unsigned int*)addr, (unsigned int)__float_as_uint(v));
}

// ---------------------------------------------------------------- utility ---
__global__ void fill_f32(float* p, float v, long n) {
    long i = (long)blockIdx.x * blockDim.x + threadIdx.x;
    long st = (long)gridDim.x * blockDim.x;
    for (; i < n; i += st) p[i] = v;
}

__global__ void f32_to_bf16(const float* __restrict__ x, bf16_t* __restrict__ y, long n) {
    long i = (long)blockIdx.x * blockDim.x + threadIdx.x;
    long st = (long)gridDim.x * blockDim.x;
    for (; i < n; i += st) y[i] = f2bf(x[i]);
}

// Repack 128x128 row-major (K x N) f32 weights into per-fragment bf16 B-layout.
// Fragment f = kt*8+nt covers K=kt*32..+31, N=nt*16..+15; lane's 16 values contiguous.
// Lane l holds column n = nt*16 + (l%16); K = kt*32 + 16*(l/16) + i, i=0..15.
__global__ void pack_w(const float* __restrict__ W, bf16_t* __restrict__ P) {
    int t = blockIdx.x * blockDim.x + threadIdx.x;
    if (t >= HID * HID) return;
    int frag = t >> 9;
    int lane = (t >> 4) & 31;
    int i    = t & 15;
    int kt = frag >> 3, nt = frag & 7;
    int k = kt * 32 + ((lane >> 4) << 4) + i;
    int n = (nt << 4) + (lane & 15);
    P[t] = f2bf(W[k * HID + n]);
}

// ---------------------------------------------------------- WMMA GEMM ------
// C[nrows x 128] = A(bf16) @ Wp(packed bf16) + bias ; one wave per 16 rows.
// Packed weights (32 KB) are staged in LDS once per 8-wave block; B fragments
// then feed v_wmma via ds_load_b128 instead of serialized global loads.
__global__ void __launch_bounds__(256) gemm_wmma(
    const bf16_t* __restrict__ A, const bf16_t* __restrict__ Wp,
    const float* __restrict__ bias, float* __restrict__ C,
    bf16_t* __restrict__ Cbf, int nrows) {
    __shared__ bf16_t Ws[HID * HID];            // 32 KB of 320 KB/WGP

    int tid = threadIdx.x;
    {   // cooperative stage: 128 B per thread (8 x b128)
        const uint4* srcp = (const uint4*)Wp;
        uint4* dstp = (uint4*)Ws;
#pragma unroll
        for (int k = 0; k < 8; ++k) dstp[tid + k * 256] = srcp[tid + k * 256];
    }
    __syncthreads();

    int gid  = blockIdx.x * blockDim.x + tid;
    int wid  = gid >> 5;
    int lane = gid & 31;
    int m0 = wid << 4;
    if (m0 + 16 > nrows) return;
    int half = lane >> 4;      // lane group 0/1
    int lm   = lane & 15;      // row within tile / col within tile

    v8f acc[8];
#pragma unroll
    for (int i = 0; i < 8; ++i)
#pragma unroll
        for (int j = 0; j < 8; ++j) acc[i][j] = 0.0f;

    const bf16_t* arow = A + (long)(m0 + lm) * HID;
#pragma unroll
    for (int kt = 0; kt < 4; ++kt) {
        // A fragment: lane holds row lm, K chunk {kt*32+16h .. +15}
        const bf16_t* ap = arow + kt * 32 + half * 16;
        v8bf lo = *(const v8bf*)(ap);
        v8bf hi = *(const v8bf*)(ap + 8);
        v16bf a = __builtin_shufflevector(lo, hi, 0, 1, 2, 3, 4, 5, 6, 7,
                                          8, 9, 10, 11, 12, 13, 14, 15);
#pragma unroll
        for (int nt = 0; nt < 8; ++nt) {
            v16bf b = *(const v16bf*)(Ws + (((kt << 3) + nt) << 9) + (lane << 4));
            acc[nt] = __builtin_amdgcn_wmma_f32_16x16x32_bf16(
                false, a, false, b, (short)0, acc[nt], false, false);
        }
    }
    // C/D layout: VGPR j -> row m0 + j + 8*half ; lane -> col nt*16 + lm
#pragma unroll
    for (int nt = 0; nt < 8; ++nt) {
        int n = (nt << 4) + lm;
        float bn = bias[n];
#pragma unroll
        for (int j = 0; j < 8; ++j) {
            long idx = (long)(m0 + (half << 3) + j) * HID + n;
            float v = acc[nt][j] + bn;
            C[idx] = v;
            if (Cbf) Cbf[idx] = f2bf(v);
        }
    }
}

// ------------------------------------------------------------ edge phase ---
// Pass 1: per (edge,head) wave -> leaky_relu(xl[src]+xr[dst]) . att ; seg max
__global__ void edge_logits(const float* __restrict__ xl, const float* __restrict__ xr,
                            const int* __restrict__ src, const int* __restrict__ dst,
                            const float* __restrict__ att, float* __restrict__ logits,
                            float* __restrict__ segmax, int E) {
    int gid  = blockIdx.x * blockDim.x + threadIdx.x;
    int lane = gid & 31;
    int wid  = gid >> 5;
    int nw   = (gridDim.x * blockDim.x) >> 5;
    int total = E * HEADS;
    for (int t = wid; t < total; t += nw) {
        int e = t >> 2, h = t & 3;
        int s = src[e], d = dst[e];
        float v = xl[(long)s * HID + h * HD + lane] + xr[(long)d * HID + h * HD + lane];
        v = v > 0.0f ? v : NEG_SLOPE * v;
        v *= att[h * HD + lane];
#pragma unroll
        for (int off = 16; off > 0; off >>= 1) v += __shfl_xor(v, off, 32);
        if (lane == 0) {
            logits[t] = v;
            atomicMaxF32(segmax + (long)d * HEADS + h, v);
        }
    }
}

// Pass 2: exp(logit - segmax[dst]) in place ; accumulate denominator
__global__ void edge_softmax_num(const int* __restrict__ dst, float* __restrict__ logits,
                                 const float* __restrict__ segmax,
                                 float* __restrict__ denom, int E) {
    int i  = blockIdx.x * blockDim.x + threadIdx.x;
    int st = gridDim.x * blockDim.x;
    int total = E * HEADS;
    for (; i < total; i += st) {
        int e = i >> 2, h = i & 3;
        int d = dst[e];
        float ex = __expf(logits[i] - segmax[(long)d * HEADS + h]);
        logits[i] = ex;
        atomicAdd(denom + (long)d * HEADS + h, ex);
    }
}

// Pass 3: accum[dst] += alpha * xl[src] ; one wave per (edge,head), lane=channel
__global__ void edge_scatter(const float* __restrict__ xl, const int* __restrict__ src,
                             const int* __restrict__ dst, const float* __restrict__ ex,
                             const float* __restrict__ denom, float* __restrict__ accum,
                             int E) {
    int gid  = blockIdx.x * blockDim.x + threadIdx.x;
    int lane = gid & 31;
    int wid  = gid >> 5;
    int nw   = (gridDim.x * blockDim.x) >> 5;
    int total = E * HEADS;
    for (int t = wid; t < total; t += nw) {
        int e = t >> 2, h = t & 3;
        int s = src[e], d = dst[e];
        float alpha = ex[t] / (denom[(long)d * HEADS + h] + 1e-16f);
        atomicAdd(accum + (long)d * HID + h * HD + lane,
                  alpha * xl[(long)s * HID + h * HD + lane]);
    }
}

// ELU + residual + LayerNorm ; one wave per node (4 channels/lane)
__global__ void post_kernel(const float* __restrict__ accum, const float* __restrict__ conv_b,
                            float* __restrict__ h, bf16_t* __restrict__ hbf,
                            const float* __restrict__ ln_g, const float* __restrict__ ln_b,
                            int N) {
    int gid  = blockIdx.x * blockDim.x + threadIdx.x;
    int lane = gid & 31;
    int n    = gid >> 5;
    if (n >= N) return;
    float vals[4];
    float s = 0.0f;
#pragma unroll
    for (int j = 0; j < 4; ++j) {
        int c = j * 32 + lane;
        float g = accum[(long)n * HID + c] + conv_b[c];
        g = g > 0.0f ? g : (__expf(g) - 1.0f);       // ELU
        g += h[(long)n * HID + c];                   // residual
        vals[j] = g;
        s += g;
    }
#pragma unroll
    for (int off = 16; off > 0; off >>= 1) s += __shfl_xor(s, off, 32);
    float mu = s * (1.0f / HID);
    float var = 0.0f;
#pragma unroll
    for (int j = 0; j < 4; ++j) { float dlt = vals[j] - mu; var += dlt * dlt; }
#pragma unroll
    for (int off = 16; off > 0; off >>= 1) var += __shfl_xor(var, off, 32);
    float inv = rsqrtf(var * (1.0f / HID) + LN_EPS);
#pragma unroll
    for (int j = 0; j < 4; ++j) {
        int c = j * 32 + lane;
        float o = (vals[j] - mu) * inv * ln_g[c] + ln_b[c];
        h[(long)n * HID + c]   = o;
        hbf[(long)n * HID + c] = f2bf(o);
    }
}

// -------------------------------------------------------------- pooling ----
__global__ void pool_acc(const float* __restrict__ h, const int* __restrict__ batch,
                         float* __restrict__ out, float* __restrict__ counts, int N) {
    int gid  = blockIdx.x * blockDim.x + threadIdx.x;
    int lane = gid & 31;
    int n    = gid >> 5;
    if (n >= N) return;
    int b = batch[n];
#pragma unroll
    for (int j = 0; j < 4; ++j) {
        int c = j * 32 + lane;
        atomicAdd(out + (long)b * HID + c, h[(long)n * HID + c]);
    }
    if (lane == 0) atomicAdd(counts + b, 1.0f);
}

__global__ void pool_div(float* out, const float* counts, int n) {
    int i = blockIdx.x * blockDim.x + threadIdx.x;
    if (i < n) out[i] /= fmaxf(counts[i >> 7], 1.0f);
}

// ================================================================ driver ===
extern "C" void kernel_launch(void* const* d_in, const int* in_sizes, int n_in,
                              void* d_out, int out_size, void* d_ws, size_t ws_size,
                              hipStream_t stream) {
    const float* x      = (const float*)d_in[0];
    const int*   edge   = (const int*)d_in[1];
    const int*   batch  = (const int*)d_in[2];
    const float* emb_W  = (const float*)d_in[3];
    const float* emb_b  = (const float*)d_in[4];
    const float* lWl    = (const float*)d_in[5];
    const float* bWl    = (const float*)d_in[6];
    const float* lWr    = (const float*)d_in[7];
    const float* bWr    = (const float*)d_in[8];
    const float* att    = (const float*)d_in[9];
    const float* conv_b = (const float*)d_in[10];
    const float* lng    = (const float*)d_in[11];
    const float* lnb    = (const float*)d_in[12];
    (void)n_in; (void)ws_size;

    const int N = in_sizes[0] / HID;
    const int E = in_sizes[1] / 2;
    const int* src = edge;
    const int* dst = edge + E;
    float* out = (float*)d_out;

    char* p = (char*)d_ws;
    auto alloc = [&](size_t bytes) -> char* {
        char* r = p;
        p += (bytes + 255) & ~(size_t)255;
        return r;
    };
    float*  h      = (float*)alloc((size_t)N * HID * 4);
    bf16_t* hbf    = (bf16_t*)alloc((size_t)N * HID * 2);
    float*  xl     = (float*)alloc((size_t)N * HID * 4);
    float*  xr     = (float*)alloc((size_t)N * HID * 4);
    float*  accum  = (float*)alloc((size_t)N * HID * 4);
    float*  logits = (float*)alloc((size_t)E * HEADS * 4);
    float*  segmax = (float*)alloc((size_t)N * HEADS * 4);
    float*  denom  = (float*)alloc((size_t)N * HEADS * 4);
    float*  counts = (float*)alloc(64 * 4);
    bf16_t* xbf    = (bf16_t*)alloc((size_t)N * HID * 2);
    bf16_t* Wp     = (bf16_t*)alloc((size_t)7 * HID * HID * 2);

    // Pack all 7 weight matrices into WMMA B-fragment order (bf16).
    pack_w<<<64, 256, 0, stream>>>(emb_W, Wp);
    for (int l = 0; l < 3; ++l) {
        pack_w<<<64, 256, 0, stream>>>(lWl + l * HID * HID, Wp + (size_t)(1 + l) * HID * HID);
        pack_w<<<64, 256, 0, stream>>>(lWr + l * HID * HID, Wp + (size_t)(4 + l) * HID * HID);
    }

    const long nx = (long)N * HID;
    f32_to_bf16<<<4096, 256, 0, stream>>>(x, xbf, nx);

    const int gw = ((N / 16) * 32 + 255) / 256;      // one wave per 16 rows
    gemm_wmma<<<gw, 256, 0, stream>>>(xbf, Wp, emb_b, h, hbf, N);

    const int egrid = 4096;                          // grid-stride wave kernels
    const int ngrid = (N * 32 + 255) / 256;          // one wave per node
    for (int l = 0; l < 3; ++l) {
        fill_f32<<<2048, 256, 0, stream>>>(accum, 0.0f, nx);
        fill_f32<<<256, 256, 0, stream>>>(segmax, -__builtin_inff(), (long)N * HEADS);
        fill_f32<<<256, 256, 0, stream>>>(denom, 0.0f, (long)N * HEADS);

        gemm_wmma<<<gw, 256, 0, stream>>>(hbf, Wp + (size_t)(1 + l) * HID * HID,
                                          bWl + l * HID, xl, (bf16_t*)nullptr, N);
        gemm_wmma<<<gw, 256, 0, stream>>>(hbf, Wp + (size_t)(4 + l) * HID * HID,
                                          bWr + l * HID, xr, (bf16_t*)nullptr, N);

        edge_logits<<<egrid, 256, 0, stream>>>(xl, xr, src, dst, att + l * HEADS * HD,
                                               logits, segmax, E);
        edge_softmax_num<<<4096, 256, 0, stream>>>(dst, logits, segmax, denom, E);
        edge_scatter<<<egrid, 256, 0, stream>>>(xl, src, dst, logits, denom, accum, E);

        post_kernel<<<ngrid, 256, 0, stream>>>(accum, conv_b + l * HID, h, hbf,
                                               lng + l * HID, lnb + l * HID, N);
    }

    fill_f32<<<8, 256, 0, stream>>>(out, 0.0f, (long)out_size);
    fill_f32<<<1, 64, 0, stream>>>(counts, 0.0f, 64);
    pool_acc<<<ngrid, 256, 0, stream>>>(h, batch, out, counts, N);
    pool_div<<<(out_size + 255) / 256, 256, 0, stream>>>(out, counts, out_size);
}
